// DrugGATNet_26671746908433
// MI455X (gfx1250) — compile-verified, matrix operand order checked by reading, not run on hardware
//
#include <hip/hip_runtime.h>
#include <hip/hip_bf16.h>

typedef __bf16 bf16;
typedef __attribute__((ext_vector_type(16))) __bf16 v16bf;
typedef __attribute__((ext_vector_type(8)))  __bf16 v8bf;
typedef __attribute__((ext_vector_type(8)))  float  v8f;

// ---------------- problem constants ----------------
constexpr int   NN    = 50000;     // nodes
constexpr int   MPAD  = 50048;     // nodes padded to 64
constexpr int   EE    = 100000;    // raw edges
constexpr int   ET    = 150000;    // edges + self loops
constexpr int   H1    = 10, C1f = 78, F1 = 780;
constexpr int   K1P   = 96;        // 78 -> pad 96 (mult 32)
constexpr int   N1COV = 896;       // 780 -> 7*128 block coverage
constexpr int   LDC1  = 784;
constexpr int   K2P   = 800;       // 780 -> pad 800
constexpr int   F2    = 128;
constexpr int   GG    = 1250;
constexpr int   GPAD  = 1280;
constexpr float NEG_SLOPE = 0.2f;
constexpr unsigned ENC_NEG_INF = 0x007FFFFFu; // enc(-inf)

// ---------------- workspace layout (bytes) ----------------
constexpr size_t alignup(size_t x) { return (x + 255) & ~(size_t)255; }
constexpr size_t oA1  = 0;
constexpr size_t oW1T = alignup(oA1  + (size_t)MPAD * K1P * 2);
constexpr size_t oAS1 = alignup(oW1T + (size_t)N1COV * K1P * 2);
constexpr size_t oAD1 = alignup(oAS1 + (size_t)NN * H1 * 4);
constexpr size_t oM1  = alignup(oAD1 + (size_t)NN * H1 * 4);
constexpr size_t oS1  = alignup(oM1  + (size_t)NN * H1 * 4);
constexpr size_t oEW1 = alignup(oS1  + (size_t)NN * H1 * 4);
constexpr size_t oR1  = alignup(oEW1 + (size_t)ET * H1 * 4);
constexpr size_t oC1  = oR1;                                   // h1 raw [MPAD,784] f32
constexpr size_t oA2  = oR1;                                   // reuse: ELU(h1) bf16 [MPAD,800]
constexpr size_t oC2  = alignup(oA2 + (size_t)MPAD * K2P * 2); // h2 raw [MPAD,128] f32
constexpr size_t oR2  = alignup(oR1 + (size_t)MPAD * LDC1 * 4);
constexpr size_t oAG1 = oR2;                                   // agg1 [NN,780] f32
constexpr size_t oW2T = oR2;                                   // reuse after agg1 dead
constexpr size_t oAS2 = alignup(oW2T + (size_t)F2 * K2P * 2);
constexpr size_t oAD2 = alignup(oAS2 + (size_t)NN * 4);
constexpr size_t oM2  = alignup(oAD2 + (size_t)NN * 4);
constexpr size_t oS2  = alignup(oM2  + (size_t)NN * 4);
constexpr size_t oEW2 = alignup(oS2  + (size_t)NN * 4);
constexpr size_t oAG2 = alignup(oEW2 + (size_t)ET * 4);
constexpr size_t oH2  = alignup(oAG2 + (size_t)NN * F2 * 4);
constexpr size_t oGE  = alignup(oH2  + (size_t)NN * F2 * 4);
constexpr size_t oGA  = alignup(oGE  + (size_t)GG * F2 * 4);
constexpr size_t oFWT = alignup(oGA  + (size_t)GPAD * F2 * 2);
constexpr size_t oC3  = alignup(oFWT + (size_t)F2 * F2 * 2);

// ---------------- helpers ----------------
__device__ __forceinline__ unsigned enc(float f) {
  unsigned u = __float_as_uint(f);
  return (u & 0x80000000u) ? ~u : (u | 0x80000000u);
}
__device__ __forceinline__ float dec(unsigned u) {
  return (u & 0x80000000u) ? __uint_as_float(u & 0x7FFFFFFFu) : __uint_as_float(~u);
}
__device__ __forceinline__ void get_edge(const long long* ei, int e, int& s, int& d) {
  if (e < EE) { s = (int)ei[e]; d = (int)ei[EE + e]; }
  else        { s = d = e - EE; }  // self loop
}

// 16-bit A/B fragment for v_wmma_*_16x16x32: lane l holds row (l&15),
// K = kh..kh+7 and 16+kh..16+kh+7 where kh = (l>>4)*8 (ISA 7.12.2 layout).
__device__ __forceinline__ v16bf load_frag(const bf16* base, int ld, int row0, int k0, int lane) {
  int r  = row0 + (lane & 15);
  int kh = (lane >> 4) << 3;
  const bf16* p = base + (size_t)r * ld + (k0 + kh);
  v8bf lo = *(const v8bf*)p;
  v8bf hi = *(const v8bf*)(p + 16);
  v16bf f;
#pragma unroll
  for (int i = 0; i < 8; ++i) { f[i] = lo[i]; f[i + 8] = hi[i]; }
  return f;
}
// f32 C/D: lane l holds column (l&15); VGPR r -> row (l>>4)*8 + r.
__device__ __forceinline__ void store_frag(float* C, int ldc, int row0, int col0, int lane,
                                           v8f c, int Nreal) {
  int n = col0 + (lane & 15);
  if (n >= Nreal) return;
  int rbase = row0 + ((lane >> 4) << 3);
#pragma unroll
  for (int r = 0; r < 8; ++r) C[(size_t)(rbase + r) * ldc + n] = c[r];
}

// ---------------- WMMA GEMM: C[M,ldc] = A[M,K](bf16) * BT[N,K]^T(bf16) ----------------
// block = 256 threads = 8 waves; wave tile 16x64; block tile 64x128.
__global__ __launch_bounds__(256) void gemm_bf16_wmma(const bf16* __restrict__ A,
                                                      const bf16* __restrict__ BT,
                                                      float* __restrict__ C,
                                                      int K, int ldc, int Nreal) {
  int lane = threadIdx.x & 31;
  int wave = threadIdx.x >> 5;
  int m0 = blockIdx.x * 64 + (wave & 3) * 16;
  int n0 = blockIdx.y * 128 + (wave >> 2) * 64;
  v8f c0 = {}, c1 = {}, c2 = {}, c3 = {};
  for (int k = 0; k < K; k += 32) {
    v16bf a  = load_frag(A,  K, m0, k, lane);
    v16bf b0 = load_frag(BT, K, n0 +  0, k, lane);
    v16bf b1 = load_frag(BT, K, n0 + 16, k, lane);
    v16bf b2 = load_frag(BT, K, n0 + 32, k, lane);
    v16bf b3 = load_frag(BT, K, n0 + 48, k, lane);
    c0 = __builtin_amdgcn_wmma_f32_16x16x32_bf16(false, a, false, b0, (short)0, c0, false, false);
    c1 = __builtin_amdgcn_wmma_f32_16x16x32_bf16(false, a, false, b1, (short)0, c1, false, false);
    c2 = __builtin_amdgcn_wmma_f32_16x16x32_bf16(false, a, false, b2, (short)0, c2, false, false);
    c3 = __builtin_amdgcn_wmma_f32_16x16x32_bf16(false, a, false, b3, (short)0, c3, false, false);
  }
  store_frag(C, ldc, m0, n0 +  0, lane, c0, Nreal);
  store_frag(C, ldc, m0, n0 + 16, lane, c1, Nreal);
  store_frag(C, ldc, m0, n0 + 32, lane, c2, Nreal);
  store_frag(C, ldc, m0, n0 + 48, lane, c3, Nreal);
}

// ---------------- packing / elementwise kernels ----------------
__global__ void fill_u32_kernel(unsigned* p, unsigned v, long long n) {
  long long t = (long long)blockIdx.x * blockDim.x + threadIdx.x;
  if (t < n) p[t] = v;
}
__global__ void pack_x_kernel(const float* __restrict__ x, bf16* __restrict__ A1) {
  long long t = (long long)blockIdx.x * blockDim.x + threadIdx.x;
  if (t >= (long long)MPAD * K1P) return;
  int row = (int)(t / K1P), k = (int)(t % K1P);
  float v = (row < NN && k < C1f) ? x[(size_t)row * C1f + k] : 0.f;
  A1[t] = (bf16)v;
}
// BT[n*Kpad + k] = W[k*Ncols + n]
__global__ void pack_wT_kernel(const float* __restrict__ W, bf16* __restrict__ BT,
                               int K, int Ncols, int Kpad, int Nrows) {
  long long t = (long long)blockIdx.x * blockDim.x + threadIdx.x;
  if (t >= (long long)Nrows * Kpad) return;
  int n = (int)(t / Kpad), k = (int)(t % Kpad);
  float v = (k < K && n < Ncols) ? W[(size_t)k * Ncols + n] : 0.f;
  BT[t] = (bf16)v;
}
// alpha_s / alpha_d per (node, head)
__global__ void alpha_kernel(const float* __restrict__ h, const float* __restrict__ asrc,
                             const float* __restrict__ adst, float* __restrict__ as,
                             float* __restrict__ ad, int Hh, int Cc, int ldh) {
  int t = blockIdx.x * blockDim.x + threadIdx.x;
  if (t >= NN * Hh) return;
  int n = t / Hh, hh = t % Hh;
  const float* hp = h + (size_t)n * ldh + hh * Cc;
  float s0 = 0.f, s1 = 0.f;
  for (int c = 0; c < Cc; ++c) {
    float hv = hp[c];
    s0 += hv * asrc[hh * Cc + c];
    s1 += hv * adst[hh * Cc + c];
  }
  as[t] = s0; ad[t] = s1;
}
__global__ void edge_max_kernel(const long long* __restrict__ ei, const float* __restrict__ as,
                                const float* __restrict__ ad, float* __restrict__ ew,
                                unsigned* __restrict__ menc, int Hh) {
  int t = blockIdx.x * blockDim.x + threadIdx.x;
  if (t >= ET * Hh) return;
  int e = t / Hh, hh = t % Hh, s, d;
  get_edge(ei, e, s, d);
  float v = as[s * Hh + hh] + ad[d * Hh + hh];
  v = v > 0.f ? v : NEG_SLOPE * v;      // leaky_relu
  ew[t] = v;
  atomicMax(&menc[d * Hh + hh], enc(v));
}
__global__ void edge_exp_kernel(const long long* __restrict__ ei, float* __restrict__ ew,
                                const unsigned* __restrict__ menc, float* __restrict__ ssum,
                                int Hh) {
  int t = blockIdx.x * blockDim.x + threadIdx.x;
  if (t >= ET * Hh) return;
  int e = t / Hh, hh = t % Hh, s, d;
  get_edge(ei, e, s, d);
  float ex = __expf(ew[t] - dec(menc[d * Hh + hh]));
  ew[t] = ex;
  atomicAdd(&ssum[d * Hh + hh], ex);
}
__global__ void edge_aggr_kernel(const long long* __restrict__ ei, const float* __restrict__ h,
                                 const float* __restrict__ ew, const float* __restrict__ ssum,
                                 float* __restrict__ outp, int Hh, int Cc, int ldh) {
  long long t = (long long)blockIdx.x * blockDim.x + threadIdx.x;
  if (t >= (long long)ET * Cc) return;
  int e = (int)(t / Cc), c = (int)(t % Cc), s, d;
  get_edge(ei, e, s, d);
  for (int hh = 0; hh < Hh; ++hh) {
    float alpha = ew[(size_t)e * Hh + hh] / (ssum[(size_t)d * Hh + hh] + 1e-16f);
    atomicAdd(&outp[(size_t)d * (Hh * Cc) + hh * Cc + c],
              h[(size_t)s * ldh + hh * Cc + c] * alpha);
  }
}
// layer1 epilogue: +b1, ELU, cast to bf16 A2 [MPAD, K2P]
__global__ void elu_pack_kernel(const float* __restrict__ agg, const float* __restrict__ b,
                                bf16* __restrict__ A2) {
  long long t = (long long)blockIdx.x * blockDim.x + threadIdx.x;
  if (t >= (long long)MPAD * K2P) return;
  int row = (int)(t / K2P), k = (int)(t % K2P);
  float v = 0.f;
  if (row < NN && k < F1) {
    v = agg[(size_t)row * F1 + k] + b[k];
    v = v > 0.f ? v : (__expf(v) - 1.f);  // elu
  }
  A2[t] = (bf16)v;
}
// layer2 epilogue: +b2, relu -> f32
__global__ void relu2_kernel(const float* __restrict__ agg, const float* __restrict__ b,
                             float* __restrict__ h2) {
  int t = blockIdx.x * blockDim.x + threadIdx.x;
  if (t >= NN * F2) return;
  float v = agg[t] + b[t % F2];
  h2[t] = v > 0.f ? v : 0.f;
}
__global__ void pool_kernel(const long long* __restrict__ batch, const float* __restrict__ h2,
                            unsigned* __restrict__ genc) {
  int t = blockIdx.x * blockDim.x + threadIdx.x;
  if (t >= NN * F2) return;
  int n = t / F2, c = t % F2;
  atomicMax(&genc[(int)batch[n] * F2 + c], enc(h2[t]));
}
__global__ void g_pack_kernel(const unsigned* __restrict__ genc, bf16* __restrict__ gA) {
  int t = blockIdx.x * blockDim.x + threadIdx.x;
  if (t >= GPAD * F2) return;
  int g = t / F2;
  float v = (g < GG) ? dec(genc[t]) : 0.f;
  gA[t] = (bf16)v;
}
__global__ void final_kernel(const float* __restrict__ C3, const float* __restrict__ fb,
                             float* __restrict__ outp) {
  int t = blockIdx.x * blockDim.x + threadIdx.x;
  if (t >= GG * F2) return;
  float v = C3[t] + fb[t % F2];
  outp[t] = v > 0.f ? v : 0.f;
}

// ---------------- launch ----------------
static inline dim3 g1(long long n) { return dim3((unsigned)((n + 255) / 256)); }

extern "C" void kernel_launch(void* const* d_in, const int* in_sizes, int n_in,
                              void* d_out, int out_size, void* d_ws, size_t ws_size,
                              hipStream_t stream) {
  (void)in_sizes; (void)n_in; (void)out_size; (void)ws_size;
  const float*     x     = (const float*)d_in[0];
  const long long* ei    = (const long long*)d_in[1];
  const long long* batch = (const long long*)d_in[2];
  const float* W1  = (const float*)d_in[3];
  const float* aS1 = (const float*)d_in[4];
  const float* aD1 = (const float*)d_in[5];
  const float* b1  = (const float*)d_in[6];
  const float* W2  = (const float*)d_in[7];
  const float* aS2 = (const float*)d_in[8];
  const float* aD2 = (const float*)d_in[9];
  const float* b2  = (const float*)d_in[10];
  const float* fcw = (const float*)d_in[11];
  const float* fcb = (const float*)d_in[12];
  float* outp = (float*)d_out;
  char*  ws   = (char*)d_ws;

  bf16*     A1   = (bf16*)(ws + oA1);
  bf16*     W1T  = (bf16*)(ws + oW1T);
  float*    as1  = (float*)(ws + oAS1);
  float*    ad1  = (float*)(ws + oAD1);
  unsigned* m1   = (unsigned*)(ws + oM1);
  float*    s1   = (float*)(ws + oS1);
  float*    ew1  = (float*)(ws + oEW1);
  float*    h1   = (float*)(ws + oC1);
  float*    ag1  = (float*)(ws + oAG1);
  bf16*     A2   = (bf16*)(ws + oA2);
  float*    h2r  = (float*)(ws + oC2);
  bf16*     W2T  = (bf16*)(ws + oW2T);
  float*    as2  = (float*)(ws + oAS2);
  float*    ad2  = (float*)(ws + oAD2);
  unsigned* m2   = (unsigned*)(ws + oM2);
  float*    s2   = (float*)(ws + oS2);
  float*    ew2  = (float*)(ws + oEW2);
  float*    ag2  = (float*)(ws + oAG2);
  float*    h2   = (float*)(ws + oH2);
  unsigned* genc = (unsigned*)(ws + oGE);
  bf16*     gA   = (bf16*)(ws + oGA);
  bf16*     fwT  = (bf16*)(ws + oFWT);
  float*    C3   = (float*)(ws + oC3);

  // ---- layer 1: h1 = x @ W1 (bf16 WMMA, f32 accumulate) ----
  pack_x_kernel<<<g1((long long)MPAD * K1P), 256, 0, stream>>>(x, A1);
  pack_wT_kernel<<<g1((long long)N1COV * K1P), 256, 0, stream>>>(W1, W1T, C1f, F1, K1P, N1COV);
  gemm_bf16_wmma<<<dim3(MPAD / 64, N1COV / 128), 256, 0, stream>>>(A1, W1T, h1, K1P, LDC1, F1);

  alpha_kernel<<<g1((long long)NN * H1), 256, 0, stream>>>(h1, aS1, aD1, as1, ad1, H1, C1f, LDC1);
  fill_u32_kernel<<<g1((long long)NN * H1), 256, 0, stream>>>(m1, ENC_NEG_INF, (long long)NN * H1);
  fill_u32_kernel<<<g1((long long)NN * H1), 256, 0, stream>>>((unsigned*)s1, 0u, (long long)NN * H1);
  fill_u32_kernel<<<g1((long long)NN * F1), 256, 0, stream>>>((unsigned*)ag1, 0u, (long long)NN * F1);
  edge_max_kernel<<<g1((long long)ET * H1), 256, 0, stream>>>(ei, as1, ad1, ew1, m1, H1);
  edge_exp_kernel<<<g1((long long)ET * H1), 256, 0, stream>>>(ei, ew1, m1, s1, H1);
  edge_aggr_kernel<<<g1((long long)ET * C1f), 256, 0, stream>>>(ei, h1, ew1, s1, ag1, H1, C1f, LDC1);
  elu_pack_kernel<<<g1((long long)MPAD * K2P), 256, 0, stream>>>(ag1, b1, A2);

  // ---- layer 2: h2 = ELU(h1) @ W2 ----
  pack_wT_kernel<<<g1((long long)F2 * K2P), 256, 0, stream>>>(W2, W2T, F1, F2, K2P, F2);
  gemm_bf16_wmma<<<dim3(MPAD / 64, 1), 256, 0, stream>>>(A2, W2T, h2r, K2P, F2, F2);

  alpha_kernel<<<g1((long long)NN), 256, 0, stream>>>(h2r, aS2, aD2, as2, ad2, 1, F2, F2);
  fill_u32_kernel<<<g1((long long)NN), 256, 0, stream>>>(m2, ENC_NEG_INF, (long long)NN);
  fill_u32_kernel<<<g1((long long)NN), 256, 0, stream>>>((unsigned*)s2, 0u, (long long)NN);
  fill_u32_kernel<<<g1((long long)NN * F2), 256, 0, stream>>>((unsigned*)ag2, 0u, (long long)NN * F2);
  edge_max_kernel<<<g1((long long)ET), 256, 0, stream>>>(ei, as2, ad2, ew2, m2, 1);
  edge_exp_kernel<<<g1((long long)ET), 256, 0, stream>>>(ei, ew2, m2, s2, 1);
  edge_aggr_kernel<<<g1((long long)ET * F2), 256, 0, stream>>>(ei, h2r, ew2, s2, ag2, 1, F2, F2);
  relu2_kernel<<<g1((long long)NN * F2), 256, 0, stream>>>(ag2, b2, h2);

  // ---- pooling + FC ----
  fill_u32_kernel<<<g1((long long)GG * F2), 256, 0, stream>>>(genc, ENC_NEG_INF, (long long)GG * F2);
  pool_kernel<<<g1((long long)NN * F2), 256, 0, stream>>>(batch, h2, genc);
  g_pack_kernel<<<g1((long long)GPAD * F2), 256, 0, stream>>>(genc, gA);
  pack_wT_kernel<<<g1((long long)F2 * F2), 256, 0, stream>>>(fcw, fwT, F2, F2, F2, F2);
  gemm_bf16_wmma<<<dim3(GPAD / 64, 1), 256, 0, stream>>>(gA, fwT, C3, F2, F2, F2);
  final_kernel<<<g1((long long)GG * F2), 256, 0, stream>>>(C3, fcb, outp);
}
